// SeiturdModel_39917426049168
// MI455X (gfx1250) — matrix operand order repaired and checked by reading, not compiled, not run on hardware
//
#include <hip/hip_runtime.h>
#include <hip/hip_bf16.h>

// MI455X / gfx1250. AI[t,r] = sum_q Io[t,q] * adj[r,q] (1023x4096 @ 4096x4096,
// 34.3 GFLOP) is the only compute-heavy op; everything else is HBM-bound
// (~320MB -> ~14us @ 23.3 TB/s). GEMM: f16-in/f32-acc v_wmma_f32_16x16x32_f16
// with block-cooperative double-buffered LDS staging via
// global_load_async_to_lds_b128 (ASYNCcnt). Round 3 refinements:
//  - all 10 fragment ds_loads issued as one clause, then 4 back-to-back WMMAs
//    (removes the per-WMMA s_wait_dscnt serialization seen in round 2)
//  - loop unrolled x2 with the tail pair peeled: stage offsets become ds_load
//    immediates, no per-iteration stage arithmetic or guard branch
//  - uniform 5 async ops per wave per stage (B chunks on lane<16 of each wave)

typedef __attribute__((ext_vector_type(16))) _Float16 v16h;
typedef __attribute__((ext_vector_type(8)))  _Float16 v8h;
typedef __attribute__((ext_vector_type(8)))  float    v8f;

#define D_   1024
#define R_   4096
#define DM1  1023
#define LOG2PI 1.8378770664093454f

#define BM 256   // block M tile (4 waves x 64 rows)
#define BN 32    // block N tile (2 waves x 16 cols)
#define BK 32    // k-step (one WMMA K)

__device__ __forceinline__ float sigmoid_f(float x) {
    return 1.0f / (1.0f + __expf(-x));
}

// k=1 multinomial->MVN logprob: cov = n p (1-p)
__device__ __forceinline__ float lp1(float n, float p, float x) {
    float mean = n * p;
    float cov  = mean * (1.0f - p);
    float d    = x - mean;
    return -0.5f * (LOG2PI + __logf(cov) + d * d / cov);
}

// k=2 multinomial->MVN logprob via closed-form 2x2 inverse (== Cholesky result)
__device__ __forceinline__ float lp2(float n, float p0, float p1, float x0, float x1) {
    float m0 = n * p0, m1 = n * p1;
    float c11 = m0 * (1.0f - p0);
    float c22 = m1 * (1.0f - p1);
    float c21 = -n * p0 * p1;
    float d0 = x0 - m0, d1 = x1 - m1;
    float det = c11 * c22 - c21 * c21;
    float quad = (c22 * d0 * d0 - 2.0f * c21 * d0 * d1 + c11 * d1 * d1) / det;
    return -0.5f * (2.0f * LOG2PI + __logf(det) + quad);
}

// ---- Kernel 1: AN^-1 per row of adjacency, and zero the output -------------
__global__ __launch_bounds__(256)
void prep_kernel(const float* __restrict__ adj, const float* __restrict__ N,
                 float* __restrict__ aninv, float* __restrict__ out) {
    __shared__ float red[256];
    int r = blockIdx.x;
    const float* row = adj + (size_t)r * R_;
    float s = 0.0f;
    for (int q = threadIdx.x; q < R_; q += 256) s += row[q] * N[q];
    red[threadIdx.x] = s;
    __syncthreads();
    for (int off = 128; off > 0; off >>= 1) {
        if (threadIdx.x < off) red[threadIdx.x] += red[threadIdx.x + off];
        __syncthreads();
    }
    if (threadIdx.x == 0) { aninv[r] = 1.0f / red[0]; out[r] = 0.0f; }
}

// ---- Kernel 2/3: fp32 -> f16 staging --------------------------------------
__global__ __launch_bounds__(256)
void cvt_adj_kernel(const float* __restrict__ adj, _Float16* __restrict__ adjh) {
    size_t i = (size_t)blockIdx.x * 256 + threadIdx.x;   // R_*R_ threads
    adjh[i] = (_Float16)adj[i];
}

__global__ __launch_bounds__(256)
void cvt_io_kernel(const float* __restrict__ I, _Float16* __restrict__ ioh) {
    size_t i = (size_t)blockIdx.x * 256 + threadIdx.x;   // D_*R_ threads
    size_t t = i >> 12;                                  // /R_
    ioh[i] = (t < DM1) ? (_Float16)I[i] : (_Float16)0.0f;
}

// ---- Kernel 4: WMMA GEMM (AI) with async-LDS double buffering -------------
// Block tile 256(M) x 32(N): 8 waves as 4(M) x 2(N), each wave 64x16 (4 accs).
// Per k-step: stage A (256x32 f16, 16KB) + B (32x32 f16, 2KB) into LDS with
// global_load_async_to_lds_b128 (ASYNCcnt), double buffered. Fragment reads
// are ds_load_b128 using the ISA 7.12.2 wave32 layouts:
//   A 16x32: lane m=lane&15; K-chunks {0..7,16..23} (+8 if lane>=16)
//   B 32x16: lane col=lane&15; K-chunks {0..7,16..23} (+8 if lane>=16)
//   C/D:     n=lane&15, m = vgpr + 8*(lane>=16)
__global__ __launch_bounds__(256)
void gemm_lps_kernel(const _Float16* __restrict__ Ah,     // f16 Io padded 1024xR
                     const _Float16* __restrict__ Bh,     // f16 adjacency RxR
                     const float* __restrict__ S,         // (D,R), rows 0..1022 used
                     const float* __restrict__ S_E,       // (D-1,R)
                     const float* __restrict__ contagion, // (D,R)
                     const float* __restrict__ aninv,     // (R)
                     float* __restrict__ out) {
    __shared__ _Float16 As[2][BM][BK];   // 32 KB
    __shared__ _Float16 Bs[2][BN][BK];   //  4 KB

    const int tid  = threadIdx.x;
    const int lane = tid & 31;
    const int wave = tid >> 5;           // 0..7
    const int wm   = wave >> 1;          // 0..3  (64-row strip)
    const int wn   = wave & 1;           // 0..1  (16-col strip)

    const int nblk = (blockIdx.x & 127) * BN;   // 128 n-blocks of 32
    const int mblk = (blockIdx.x >> 7) * BM;    //   4 m-blocks of 256

    const int mlo = lane & 15;
    const int hi  = lane >> 4;

    // One k-stage of async copies; exactly 5 async instructions per wave:
    // A = 1024 16B chunks (4/thread), B = 128 16B chunks (lane<16 of each wave).
    auto stage_load = [&](int st, int k0) {
        #pragma unroll
        for (int j = 0; j < 4; ++j) {
            int ci   = tid + 256 * j;          // 0..1023
            int row  = ci >> 2;                // 0..255
            int part = ci & 3;                 // 16B chunk within 64B row
            unsigned long long ga =
                (unsigned long long)(Ah + (size_t)(mblk + row) * R_ + k0 + part * 8);
            unsigned ldso = (unsigned)(size_t)&As[st][row][part * 8];
            asm volatile("global_load_async_to_lds_b128 %0, %1, off"
                         :: "v"(ldso), "v"(ga) : "memory");
        }
        if (lane < 16) {
            int ci   = wave * 16 + lane;       // 0..127
            int row  = ci >> 2;                // 0..31
            int part = ci & 3;
            unsigned long long ga =
                (unsigned long long)(Bh + (size_t)(nblk + row) * R_ + k0 + part * 8);
            unsigned ldso = (unsigned)(size_t)&Bs[st][row][part * 8];
            asm volatile("global_load_async_to_lds_b128 %0, %1, off"
                         :: "v"(ldso), "v"(ga) : "memory");
        }
    };

    v8f acc0 = {}, acc1 = {}, acc2 = {}, acc3 = {};

    // Per-wave LDS fragment bases (stage/subtile offsets fold into ds offsets).
    const _Float16* aBase = &As[0][wm * 64 + mlo][hi * 8];
    const _Float16* bBase = &Bs[0][wn * 16 + mlo][hi * 8];

    // All 10 ds_loads issued together, one dscnt wait, then 4 WMMAs.
    auto compute = [&](int st) {
        const _Float16* ab = aBase + st * (BM * BK);   // +8192 halves
        const _Float16* bb = bBase + st * (BN * BK);   // +1024 halves
        v8h b0  = *(const v8h*)bb;
        v8h b1  = *(const v8h*)(bb + 16);
        v8h a00 = *(const v8h*)(ab);
        v8h a01 = *(const v8h*)(ab + 16);
        v8h a10 = *(const v8h*)(ab + 512);
        v8h a11 = *(const v8h*)(ab + 528);
        v8h a20 = *(const v8h*)(ab + 1024);
        v8h a21 = *(const v8h*)(ab + 1040);
        v8h a30 = *(const v8h*)(ab + 1536);
        v8h a31 = *(const v8h*)(ab + 1552);
        v16h b = __builtin_shufflevector(b0, b1, 0,1,2,3,4,5,6,7,8,9,10,11,12,13,14,15);
        v16h a = __builtin_shufflevector(a00, a01, 0,1,2,3,4,5,6,7,8,9,10,11,12,13,14,15);
        acc0 = __builtin_amdgcn_wmma_f32_16x16x32_f16(false, a, false, b, (short)0, acc0, false, false);
        a = __builtin_shufflevector(a10, a11, 0,1,2,3,4,5,6,7,8,9,10,11,12,13,14,15);
        acc1 = __builtin_amdgcn_wmma_f32_16x16x32_f16(false, a, false, b, (short)0, acc1, false, false);
        a = __builtin_shufflevector(a20, a21, 0,1,2,3,4,5,6,7,8,9,10,11,12,13,14,15);
        acc2 = __builtin_amdgcn_wmma_f32_16x16x32_f16(false, a, false, b, (short)0, acc2, false, false);
        a = __builtin_shufflevector(a30, a31, 0,1,2,3,4,5,6,7,8,9,10,11,12,13,14,15);
        acc3 = __builtin_amdgcn_wmma_f32_16x16x32_f16(false, a, false, b, (short)0, acc3, false, false);
    };

    stage_load(0, 0);
    asm volatile("s_wait_asynccnt 0x0" ::: "memory");
    __syncthreads();

    int k0 = 0;
    for (; k0 < R_ - 2 * BK; k0 += 2 * BK) {
        stage_load(1, k0 + BK);        // prefetch next stage (overlaps compute)
        compute(0);
        asm volatile("s_wait_asynccnt 0x0" ::: "memory");
        __syncthreads();

        stage_load(0, k0 + 2 * BK);
        compute(1);
        asm volatile("s_wait_asynccnt 0x0" ::: "memory");
        __syncthreads();
    }
    // Tail pair: k0 == R_ - 64
    stage_load(1, k0 + BK);
    compute(0);
    asm volatile("s_wait_asynccnt 0x0" ::: "memory");
    __syncthreads();
    compute(1);

    // Fused epilogue: p_SE = -expm1(-(c * AI) / AN); k=1 logprob; sum over t.
    const int nbase = nblk + wn * 16;
    const int mbase = mblk + wm * 64;
    const int r = nbase + mlo;
    const float ai_inv = aninv[r];
    float lsum = 0.0f;

    v8f accs[4] = {acc0, acc1, acc2, acc3};
    #pragma unroll
    for (int s = 0; s < 4; ++s) {
        int tb = mbase + s * 16 + hi * 8;
        #pragma unroll
        for (int v = 0; v < 8; ++v) {
            int t = tb + v;
            if (t < DM1) {
                size_t idx = (size_t)t * R_ + r;
                float p = -expm1f(-(contagion[idx] * accs[s][v]) * ai_inv);
                lsum += lp1(S[idx], p, S_E[idx]);
            }
        }
    }
    __hip_atomic_fetch_add(&out[r], lsum, __ATOMIC_RELAXED, __HIP_MEMORY_SCOPE_AGENT);
}

// ---- Kernel 5: elementwise lpE + lpI + lpT, reduced over t ----------------
__global__ __launch_bounds__(256)
void elem_lp_kernel(const float* __restrict__ E,  const float* __restrict__ I,
                    const float* __restrict__ T,
                    const float* __restrict__ E_I, const float* __restrict__ I_T,
                    const float* __restrict__ I_U, const float* __restrict__ T_R,
                    const float* __restrict__ T_D,
                    const float* __restrict__ detr, const float* __restrict__ recr,
                    const float* __restrict__ ldE, const float* __restrict__ ldI,
                    const float* __restrict__ ldT,
                    float* __restrict__ out) {
    int r  = blockIdx.x * 256 + threadIdx.x;       // gridDim.x = 16
    int t0 = blockIdx.y * 32;                      // gridDim.y = 32
    int t1 = t0 + 32; if (t1 > DM1) t1 = DM1;

    float dE = sigmoid_f(ldE[0]);
    float dI = sigmoid_f(ldI[0]);
    float dT = sigmoid_f(ldT[0]);

    float sum = 0.0f;
    for (int t = t0; t < t1; ++t) {
        size_t idx = (size_t)t * R_ + r;
        sum += lp1(E[idx], dE, E_I[idx]);
        float de = detr[idx];
        sum += lp2(I[idx], de * dI, (1.0f - de) * dI, I_T[idx], I_U[idx]);
        float rc = recr[idx];
        sum += lp2(T[idx], rc * dT, (1.0f - rc) * dT, T_R[idx], T_D[idx]);
    }
    __hip_atomic_fetch_add(&out[r], sum, __ATOMIC_RELAXED, __HIP_MEMORY_SCOPE_AGENT);
}

extern "C" void kernel_launch(void* const* d_in, const int* in_sizes, int n_in,
                              void* d_out, int out_size, void* d_ws, size_t ws_size,
                              hipStream_t stream) {
    const float* S    = (const float*)d_in[0];
    const float* E    = (const float*)d_in[1];
    const float* I    = (const float*)d_in[2];
    const float* T    = (const float*)d_in[3];
    const float* N    = (const float*)d_in[4];
    const float* S_E  = (const float*)d_in[5];
    const float* E_I  = (const float*)d_in[6];
    const float* I_T  = (const float*)d_in[7];
    const float* I_U  = (const float*)d_in[8];
    const float* T_R  = (const float*)d_in[9];
    const float* T_D  = (const float*)d_in[10];
    const float* adj  = (const float*)d_in[11];
    const float* ldE  = (const float*)d_in[12];
    const float* ldI  = (const float*)d_in[13];
    const float* ldT  = (const float*)d_in[14];
    const float* detr = (const float*)d_in[15];
    const float* recr = (const float*)d_in[16];
    const float* cont = (const float*)d_in[17];
    float* out = (float*)d_out;

    // Workspace layout: f16 adjacency (32MB) | f16 Io padded (8MB) | AN^-1 (16KB)
    char* ws = (char*)d_ws;
    _Float16* adjh  = (_Float16*)ws;
    _Float16* ioh   = (_Float16*)(ws + (size_t)R_ * R_ * sizeof(_Float16));
    float*    aninv = (float*)(ws + (size_t)R_ * R_ * sizeof(_Float16)
                                  + (size_t)D_ * R_ * sizeof(_Float16));

    prep_kernel   <<<R_, 256, 0, stream>>>(adj, N, aninv, out);
    cvt_adj_kernel<<<(R_ * R_) / 256, 256, 0, stream>>>(adj, adjh);
    cvt_io_kernel <<<(D_ * R_) / 256, 256, 0, stream>>>(I, ioh);
    // 512 blocks: (1024/256 M-blocks) x (4096/32 N-blocks), 8 waves/block
    gemm_lps_kernel<<<512, 256, 0, stream>>>(ioh, adjh, S, S_E, cont, aninv, out);
    elem_lp_kernel<<<dim3(16, 32), 256, 0, stream>>>(E, I, T, E_I, I_T, I_U,
                                                     T_R, T_D, detr, recr,
                                                     ldE, ldI, ldT, out);
}
